// get_model_45956150067276
// MI455X (gfx1250) — matrix operand order repaired and checked by reading, not compiled, not run on hardware
//
#include <hip/hip_runtime.h>
#include <stdint.h>

// ---------------------------------------------------------------------------
// PointNet++ MSG forward for gfx1250 (CDNA5, wave32).
// Shared-MLP GEMMs run on v_wmma_f32_16x16x32_f16 (f16 in, f32 acc) with
// weights pre-packed in B-fragment order. Grouping+MLP+maxpool fused in LDS.
// ---------------------------------------------------------------------------

#define BN_SCALE_C 0.99999500003749968f  // 1/sqrt(1+1e-5)

typedef _Float16 h8   __attribute__((ext_vector_type(8)));
typedef _Float16 v16h __attribute__((ext_vector_type(16)));
typedef float    v8f  __attribute__((ext_vector_type(8)));

struct WLayer {
    const _Float16* Wt;   // packed B-fragment layout, O*Cpad halves
    const float*    sc;   // BN_SCALE*g
    const float*    bi;   // b*sc + beta
    int Cpad;             // input channels, multiple of 32
    int O;                // output channels, multiple of 16
};
struct Mlp3 { WLayer l[3]; };

// ---- WMMA fragment loaders (layouts per CDNA5 ISA 7.12.2) ------

// A (16x32, f16): lane<16 row M=lane holds K = {kBase+0..7, kBase+16..23};
// lane>=16 row M=lane-16 holds K = {kBase+8..15, kBase+24..31}.
__device__ __forceinline__ v16h load_a_frag(const _Float16* base, int ldh,
                                            int mBase, int kBase, int lane) {
    int half = lane >> 4;
    int m = mBase + (lane & 15);
    const _Float16* row = base + (size_t)m * ldh + kBase + half * 8;
    h8 lo = *(const h8*)(row);        // K = kBase + half*8 + 0..7
    h8 hi = *(const h8*)(row + 16);   // K = kBase + 16 + half*8 + 0..7
    return __builtin_shufflevector(lo, hi, 0,1,2,3,4,5,6,7,8,9,10,11,12,13,14,15);
}

// B (32x16, f16) pre-packed: lane-contiguous 16 halves at
// (((tn*(Cpad/32)+kc)*32 + lane) * 16).
__device__ __forceinline__ v16h load_b_frag(const _Float16* Wt, int Cpad,
                                            int tn, int kc, int lane) {
    const _Float16* p = Wt + ((((size_t)tn * (Cpad >> 5) + kc) * 32 + lane) << 4);
    h8 lo = *(const h8*)(p);
    h8 hi = *(const h8*)(p + 8);
    return __builtin_shufflevector(lo, hi, 0,1,2,3,4,5,6,7,8,9,10,11,12,13,14,15);
}

__device__ __forceinline__ v8f wmma16(v16h a, v16h b, v8f c) {
    return __builtin_amdgcn_wmma_f32_16x16x32_f16(false, a, false, b,
                                                  (short)0, c, false, false);
}

// ---- misc prep kernels -----------------------------------------------------

// xyz (B,6,N) -> pts f32 (B,N,3), norm f16 (B,N,3)
__global__ __launch_bounds__(256) void k_split(const float* __restrict__ xyz,
                                               float* __restrict__ pts,
                                               _Float16* __restrict__ normh,
                                               int B, int N) {
    int t = blockIdx.x * blockDim.x + threadIdx.x;
    if (t >= B * N * 3) return;
    int c = t % 3, n = (t / 3) % N, b = t / (3 * N);
    pts[t]   = xyz[((size_t)b * 6 + c) * N + n];
    normh[t] = (_Float16)xyz[((size_t)b * 6 + 3 + c) * N + n];
}

// W (O x C) f32 -> packed f16 B-fragment layout (zero-pad k>=C)
__global__ __launch_bounds__(256) void k_prep_w(const float* __restrict__ W,
                                                int O, int C, int Cpad,
                                                _Float16* __restrict__ Wt) {
    int t = blockIdx.x * blockDim.x + threadIdx.x;
    if (t >= O * Cpad) return;
    int j = t & 15, lane = (t >> 4) & 31, rest = t >> 9;
    int nkc = Cpad >> 5;
    int kc = rest % nkc, tileN = rest / nkc;
    int n = tileN * 16 + (lane & 15);
    int k = kc * 32 + ((j >> 3) << 4) + ((lane >> 4) << 3) + (j & 7);
    _Float16 v = (_Float16)0.f;
    if (k < C) v = (_Float16)W[(size_t)n * C + k];
    Wt[t] = v;
}

__global__ __launch_bounds__(256) void k_prep_sb(const float* __restrict__ b,
                                                 const float* __restrict__ g,
                                                 const float* __restrict__ beta,
                                                 int O, float* __restrict__ sc,
                                                 float* __restrict__ bi) {
    int t = blockIdx.x * blockDim.x + threadIdx.x;
    if (t >= O) return;
    float s = (float)BN_SCALE_C * g[t];
    sc[t] = s;
    bi[t] = b[t] * s + beta[t];
}

// ---- farthest point sampling (1 block per batch, sequential scan) ---------
__global__ __launch_bounds__(256) void k_fps(const float* __restrict__ xyz,
                                             int N, int npoint,
                                             float* __restrict__ new_xyz) {
    __shared__ float dist[1024];
    __shared__ float rmax[256];
    __shared__ int   ridx[256];
    int b = blockIdx.x, tid = threadIdx.x;
    const float* px = xyz + (size_t)b * N * 3;
    for (int i = tid; i < N; i += blockDim.x) dist[i] = 1e10f;
    __syncthreads();
    int far = 0;
    for (int s = 0; s < npoint; ++s) {
        float cx = px[far * 3 + 0], cy = px[far * 3 + 1], cz = px[far * 3 + 2];
        if (tid == 0) {
            float* o = new_xyz + ((size_t)b * npoint + s) * 3;
            o[0] = cx; o[1] = cy; o[2] = cz;
        }
        float bm = -1.f; int bi = 0;
        for (int i = tid; i < N; i += blockDim.x) {
            float dx = px[i * 3 + 0] - cx;
            float dy = px[i * 3 + 1] - cy;
            float dz = px[i * 3 + 2] - cz;
            float nd = fminf(dist[i], dx * dx + dy * dy + dz * dz);
            dist[i] = nd;
            if (nd > bm) { bm = nd; bi = i; }
        }
        rmax[tid] = bm; ridx[tid] = bi;
        __syncthreads();
        for (int off = blockDim.x >> 1; off > 0; off >>= 1) {
            if (tid < off) {
                if (rmax[tid + off] > rmax[tid] ||
                    (rmax[tid + off] == rmax[tid] && ridx[tid + off] < ridx[tid])) {
                    rmax[tid] = rmax[tid + off];
                    ridx[tid] = ridx[tid + off];
                }
            }
            __syncthreads();
        }
        far = ridx[0];
        __syncthreads();
    }
}

// ---- ball query: first K indices (ascending) within radius, pad w/ first --
__global__ __launch_bounds__(256) void k_ball(const float* __restrict__ xyz, int N,
                                              const float* __restrict__ new_xyz,
                                              int S, float r2, int K,
                                              int* __restrict__ idx, int total) {
    int t = blockIdx.x * blockDim.x + threadIdx.x;
    if (t >= total) return;
    int b = t / S, s = t % S;
    const float* nx = new_xyz + ((size_t)b * S + s) * 3;
    float cx = nx[0], cy = nx[1], cz = nx[2];
    int* out = idx + ((size_t)b * S + s) * K;
    int cnt = 0, first = 0;
    for (int i = 0; i < N && cnt < K; ++i) {
        const float* p = xyz + ((size_t)b * N + i) * 3;
        float dx = p[0] - cx, dy = p[1] - cy, dz = p[2] - cz;
        if (dx * dx + dy * dy + dz * dz <= r2) {
            if (cnt == 0) first = i;
            out[cnt++] = i;
        }
    }
    for (; cnt < K; ++cnt) out[cnt] = first;
}

// ---- fused: gather group -> 3x (WMMA GEMM + BN + ReLU) -> maxpool over K ---
// One block per (b,s) group; everything stays in (dynamic) LDS.
__global__ __launch_bounds__(256) void k_sa_branch(
        const float* __restrict__ xyz, int N,
        const _Float16* __restrict__ feat, int Cf, int ldF,
        const float* __restrict__ centers,
        const int* __restrict__ nidx,
        int S, int K, Mlp3 mlp,
        _Float16* __restrict__ outFeat, int outOff, int ldOut) {
    extern __shared__ _Float16 smem[];
    int b = blockIdx.x / S, s = blockIdx.x % S;
    int tid = threadIdx.x;

    int C0 = mlp.l[0].Cpad;
    int r0 = K * (C0 > mlp.l[1].O ? C0 : mlp.l[1].O);
    _Float16* bufA = smem;
    _Float16* bufB = smem + r0;

    // gather: rows = K neighbors, cols = [feat(Cf) | rel_xyz(3) | zero-pad]
    const float* ctr = centers + ((size_t)b * S + s) * 3;
    float cx = ctr[0], cy = ctr[1], cz = ctr[2];
    const int* gi = nidx + ((size_t)b * S + s) * K;
    for (int t = tid; t < K * C0; t += blockDim.x) {
        int k = t / C0, c = t % C0;
        int pi = gi[k];
        _Float16 v = (_Float16)0.f;
        if (c < Cf) {
            v = feat[((size_t)b * N + pi) * ldF + c];
        } else if (c < Cf + 3) {
            int d = c - Cf;
            float q = xyz[((size_t)b * N + pi) * 3 + d] -
                      (d == 0 ? cx : (d == 1 ? cy : cz));
            v = (_Float16)q;
        }
        bufA[(size_t)k * C0 + c] = v;
    }
    __syncthreads();

    int wave = tid >> 5, lane = tid & 31, nW = blockDim.x >> 5;
    _Float16* cur = bufA;
    _Float16* nxt = bufB;
    for (int li = 0; li < 3; ++li) {
        WLayer L = mlp.l[li];
        int tilesN = L.O >> 4;
        int nT = (K >> 4) * tilesN;
        int nkc = L.Cpad >> 5;
        for (int t = wave; t < nT; t += nW) {
            int tm = t / tilesN, tn = t % tilesN;
            v8f acc = {};
            for (int kc = 0; kc < nkc; ++kc) {
                v16h a = load_a_frag(cur, L.Cpad, tm << 4, kc << 5, lane);
                v16h bb = load_b_frag(L.Wt, L.Cpad, tn, kc, lane);
                acc = wmma16(a, bb, acc);
            }
            int n = (tn << 4) + (lane & 15);
            float sc = L.sc[n], bi = L.bi[n];
            int mBase = (tm << 4) + ((lane >> 4) ? 8 : 0);
#pragma unroll
            for (int i = 0; i < 8; ++i) {
                float v = acc[i] * sc + bi;
                v = v > 0.f ? v : 0.f;
                nxt[(size_t)(mBase + i) * L.O + n] = (_Float16)v;
            }
        }
        __syncthreads();
        _Float16* tmp = cur; cur = nxt; nxt = tmp;
    }

    // maxpool over K rows -> out features (f16 for next stage)
    int O2 = mlp.l[2].O;
    for (int o = tid; o < O2; o += blockDim.x) {
        float m = -3.4e38f;
        for (int k = 0; k < K; ++k)
            m = fmaxf(m, (float)cur[(size_t)k * O2 + o]);
        outFeat[((size_t)b * S + s) * ldOut + outOff + o] = (_Float16)m;
    }
}

// ---- flat global-memory WMMA GEMM (SA3): Y = relu(X*Wt*sc+bi) --------------
__global__ __launch_bounds__(256) void k_gemm(const _Float16* __restrict__ X,
                                              int M, WLayer L,
                                              _Float16* __restrict__ Y) {
    int wave = threadIdx.x >> 5, lane = threadIdx.x & 31;
    int tilesN = L.O >> 4;
    int nT = (M >> 4) * tilesN;
    int t = blockIdx.x * (int)(blockDim.x >> 5) + wave;   // wave-uniform
    if (t >= nT) return;                                  // whole wave exits
    int tm = t / tilesN, tn = t % tilesN;
    int nkc = L.Cpad >> 5;
    v8f acc = {};
    for (int kc = 0; kc < nkc; ++kc) {
        const _Float16* wp =
            L.Wt + ((((size_t)tn * nkc + kc) * 32 + lane) << 4);
        if (kc + 1 < nkc) __builtin_prefetch(wp + 512, 0, 0);
        v16h a = load_a_frag(X, L.Cpad, tm << 4, kc << 5, lane);
        h8 lo = *(const h8*)(wp);
        h8 hi = *(const h8*)(wp + 8);
        v16h bb = __builtin_shufflevector(lo, hi, 0,1,2,3,4,5,6,7,8,9,10,11,12,13,14,15);
        acc = wmma16(a, bb, acc);
    }
    int n = (tn << 4) + (lane & 15);
    float sc = L.sc[n], bi = L.bi[n];
    int mBase = (tm << 4) + ((lane >> 4) ? 8 : 0);
#pragma unroll
    for (int i = 0; i < 8; ++i) {
        float v = acc[i] * sc + bi;
        v = v > 0.f ? v : 0.f;
        Y[(size_t)(mBase + i) * L.O + n] = (_Float16)v;
    }
}

// ---- SA3 input: rows=B*128, cols = [xyz(3) | l2_pts(640) | pad -> 672] ----
__global__ __launch_bounds__(256) void k_build_x3(const float* __restrict__ l2x,
                                                  const _Float16* __restrict__ l2p,
                                                  _Float16* __restrict__ X) {
    int t = blockIdx.x * blockDim.x + threadIdx.x;
    if (t >= 2048 * 672) return;
    int r = t / 672, c = t % 672;
    _Float16 v = (_Float16)0.f;
    if (c < 3)        v = (_Float16)l2x[(size_t)r * 3 + c];
    else if (c < 643) v = l2p[(size_t)r * 640 + (c - 3)];
    X[t] = v;
}

__global__ __launch_bounds__(256) void k_maxpool3(const _Float16* __restrict__ Y,
                                                  float* __restrict__ out,
                                                  int B, int K, int C) {
    int t = blockIdx.x * blockDim.x + threadIdx.x;
    if (t >= B * C) return;
    int b = t / C, o = t % C;
    float m = -3.4e38f;
    for (int k = 0; k < K; ++k)
        m = fmaxf(m, (float)Y[((size_t)b * K + k) * C + o]);
    out[t] = m;
}

// ---- FC head (tiny, scalar f32) -------------------------------------------
__global__ __launch_bounds__(256) void k_fc(const float* __restrict__ X, int M,
                                            int Cin, const float* __restrict__ W,
                                            const float* __restrict__ b,
                                            const float* __restrict__ g,
                                            const float* __restrict__ beta,
                                            int useBn, int O,
                                            float* __restrict__ Y) {
    int t = blockIdx.x * blockDim.x + threadIdx.x;
    if (t >= M * O) return;
    int m = t / O, o = t % O;
    const float* xr = X + (size_t)m * Cin;
    const float* wr = W + (size_t)o * Cin;
    float acc = 0.f;
    for (int c = 0; c < Cin; ++c) acc += xr[c] * wr[c];
    acc += b[o];
    if (useBn) {
        acc = acc * ((float)BN_SCALE_C * g[o]) + beta[o];
        acc = fmaxf(acc, 0.f);
    }
    Y[t] = acc;
}

__global__ __launch_bounds__(64) void k_logsoftmax(const float* __restrict__ X,
                                                   float* __restrict__ out,
                                                   int M, int C) {
    int m = blockIdx.x * blockDim.x + threadIdx.x;
    if (m >= M) return;
    const float* r = X + (size_t)m * C;
    float mx = r[0];
    for (int j = 1; j < C; ++j) mx = fmaxf(mx, r[j]);
    float s = 0.f;
    for (int j = 0; j < C; ++j) s += __expf(r[j] - mx);
    float l = __logf(s);
    for (int j = 0; j < C; ++j) out[(size_t)m * C + j] = r[j] - mx - l;
}

// ---------------------------------------------------------------------------
extern "C" void kernel_launch(void* const* d_in, const int* in_sizes, int n_in,
                              void* d_out, int out_size, void* d_ws, size_t ws_size,
                              hipStream_t stream) {
    (void)in_sizes; (void)n_in; (void)out_size; (void)ws_size;
    const int B = 16, N1 = 1024, S1 = 512, S2 = 128, NC = 40;
    const float* xyz = (const float*)d_in[0];

    // --- walk params in setup_inputs() insertion order ---
    struct LayerCfg { int C, Cpad, O; const float *W, *b, *g, *beta; };
    LayerCfg L[21]; int nl = 0; int p = 1;
    auto padC = [](int c) { return (c + 31) / 32 * 32; };
    const int sa1w[3][3] = {{32,32,64},{64,64,128},{64,96,128}};
    const int sa2w[3][3] = {{64,64,128},{128,128,256},{128,128,256}};
    const int sa3w[3]    = {256,512,1024};
    for (int br = 0; br < 3; ++br) {
        int cin = 6;
        for (int li = 0; li < 3; ++li) {
            LayerCfg c; c.C = cin; c.Cpad = padC(cin); c.O = sa1w[br][li];
            c.W = (const float*)d_in[p++]; c.b = (const float*)d_in[p++];
            c.g = (const float*)d_in[p++]; c.beta = (const float*)d_in[p++];
            L[nl++] = c; cin = c.O;
        }
    }
    for (int br = 0; br < 3; ++br) {
        int cin = 323;
        for (int li = 0; li < 3; ++li) {
            LayerCfg c; c.C = cin; c.Cpad = padC(cin); c.O = sa2w[br][li];
            c.W = (const float*)d_in[p++]; c.b = (const float*)d_in[p++];
            c.g = (const float*)d_in[p++]; c.beta = (const float*)d_in[p++];
            L[nl++] = c; cin = c.O;
        }
    }
    {
        int cin = 643;
        for (int li = 0; li < 3; ++li) {
            LayerCfg c; c.C = cin; c.Cpad = padC(cin); c.O = sa3w[li];
            c.W = (const float*)d_in[p++]; c.b = (const float*)d_in[p++];
            c.g = (const float*)d_in[p++]; c.beta = (const float*)d_in[p++];
            L[nl++] = c; cin = c.O;
        }
    }
    const float* fc1W = (const float*)d_in[p++]; const float* fc1b = (const float*)d_in[p++];
    const float* bn1g = (const float*)d_in[p++]; const float* bn1be = (const float*)d_in[p++];
    const float* fc2W = (const float*)d_in[p++]; const float* fc2b = (const float*)d_in[p++];
    const float* bn2g = (const float*)d_in[p++]; const float* bn2be = (const float*)d_in[p++];
    const float* fc3W = (const float*)d_in[p++]; const float* fc3b = (const float*)d_in[p++];

    // --- workspace carve (bump allocator, ~22 MB total) ---
    char* wsb = (char*)d_ws; size_t off = 0;
    auto alloc = [&](size_t bytes) -> void* {
        void* r = wsb + off;
        off = (off + bytes + 255) & ~(size_t)255;
        return r;
    };
    float*     pts   = (float*)alloc((size_t)B * N1 * 3 * 4);
    _Float16*  normh = (_Float16*)alloc((size_t)B * N1 * 3 * 2);
    float*     l1x   = (float*)alloc((size_t)B * S1 * 3 * 4);
    _Float16*  l1p   = (_Float16*)alloc((size_t)B * S1 * 320 * 2);
    float*     l2x   = (float*)alloc((size_t)B * S2 * 3 * 4);
    _Float16*  l2p   = (_Float16*)alloc((size_t)B * S2 * 640 * 2);
    int*       idxb  = (int*)alloc((size_t)B * S1 * 128 * 4);
    _Float16*  X3a   = (_Float16*)alloc((size_t)B * S2 * 672 * 2);
    _Float16*  X3b   = (_Float16*)alloc((size_t)B * S2 * 1024 * 2);
    float*     gfeat = (float*)alloc((size_t)B * 1024 * 4);
    float*     f1    = (float*)alloc((size_t)B * 512 * 4);
    float*     f2    = (float*)alloc((size_t)B * 256 * 4);
    float*     lg    = (float*)alloc((size_t)B * NC * 4);

    WLayer dl[21];
    for (int i = 0; i < 21; ++i) {
        dl[i].Wt   = (_Float16*)alloc((size_t)L[i].O * L[i].Cpad * 2);
        dl[i].sc   = (float*)alloc((size_t)L[i].O * 4);
        dl[i].bi   = (float*)alloc((size_t)L[i].O * 4);
        dl[i].Cpad = L[i].Cpad;
        dl[i].O    = L[i].O;
    }

    (void)hipFuncSetAttribute(reinterpret_cast<const void*>(k_sa_branch),
                              hipFuncAttributeMaxDynamicSharedMemorySize, 160 * 1024);

    // --- weight packing + BN fold ---
    for (int i = 0; i < 21; ++i) {
        int tot = L[i].O * L[i].Cpad;
        k_prep_w<<<(tot + 255) / 256, 256, 0, stream>>>(
            L[i].W, L[i].O, L[i].C, L[i].Cpad, (_Float16*)dl[i].Wt);
        k_prep_sb<<<(L[i].O + 255) / 256, 256, 0, stream>>>(
            L[i].b, L[i].g, L[i].beta, L[i].O, (float*)dl[i].sc, (float*)dl[i].bi);
    }

    // --- split / transpose input ---
    k_split<<<(B * N1 * 3 + 255) / 256, 256, 0, stream>>>(xyz, pts, normh, B, N1);

    auto ldsBytes = [](int K, const Mlp3& m) -> size_t {
        int r0 = K * (m.l[0].Cpad > m.l[1].O ? m.l[0].Cpad : m.l[1].O);
        int r1 = K * (m.l[0].O > m.l[2].O ? m.l[0].O : m.l[2].O);
        return (size_t)(r0 + r1) * 2;
    };

    // --- SA1 ---
    k_fps<<<B, 256, 0, stream>>>(pts, N1, S1, l1x);
    {
        const float rv[3] = {0.1f, 0.2f, 0.4f};
        const int   kv[3] = {16, 32, 128};
        const int   ofv[3] = {0, 64, 192};
        for (int br = 0; br < 3; ++br) {
            int K = kv[br];
            k_ball<<<(B * S1 + 255) / 256, 256, 0, stream>>>(
                pts, N1, l1x, S1, rv[br] * rv[br], K, idxb, B * S1);
            Mlp3 m = { { dl[br * 3 + 0], dl[br * 3 + 1], dl[br * 3 + 2] } };
            k_sa_branch<<<B * S1, 256, ldsBytes(K, m), stream>>>(
                pts, N1, normh, 3, 3, l1x, idxb, S1, K, m, l1p, ofv[br], 320);
        }
    }

    // --- SA2 ---
    k_fps<<<B, 256, 0, stream>>>(l1x, S1, S2, l2x);
    {
        const float rv[3] = {0.2f, 0.4f, 0.8f};
        const int   kv[3] = {32, 64, 128};
        const int   ofv[3] = {0, 128, 384};
        for (int br = 0; br < 3; ++br) {
            int K = kv[br];
            k_ball<<<(B * S2 + 255) / 256, 256, 0, stream>>>(
                l1x, S1, l2x, S2, rv[br] * rv[br], K, idxb, B * S2);
            Mlp3 m = { { dl[9 + br * 3 + 0], dl[9 + br * 3 + 1], dl[9 + br * 3 + 2] } };
            k_sa_branch<<<B * S2, 256, ldsBytes(K, m), stream>>>(
                l1x, S1, l1p, 320, 320, l2x, idxb, S2, K, m, l2p, ofv[br], 640);
        }
    }

    // --- SA3 (flat WMMA GEMMs, M = B*128 = 2048) ---
    k_build_x3<<<(2048 * 672 + 255) / 256, 256, 0, stream>>>(l2x, l2p, X3a);
    {
        int M = B * S2;
        const _Float16* src[3] = { X3a, X3b, X3a };
        _Float16*       dst[3] = { X3b, X3a, X3b };
        for (int li = 0; li < 3; ++li) {
            WLayer Lw = dl[18 + li];
            int nT = (M >> 4) * (Lw.O >> 4);
            k_gemm<<<(nT + 7) / 8, 256, 0, stream>>>(src[li], M, Lw, dst[li]);
        }
    }
    k_maxpool3<<<(B * 1024 + 255) / 256, 256, 0, stream>>>(X3b, gfeat, B, S2, 1024);

    // --- FC head + log_softmax ---
    k_fc<<<(B * 512 + 255) / 256, 256, 0, stream>>>(gfeat, B, 1024, fc1W, fc1b,
                                                    bn1g, bn1be, 1, 512, f1);
    k_fc<<<(B * 256 + 255) / 256, 256, 0, stream>>>(f1, B, 512, fc2W, fc2b,
                                                    bn2g, bn2be, 1, 256, f2);
    k_fc<<<(B * NC + 255) / 256, 256, 0, stream>>>(f2, B, 256, fc3W, fc3b,
                                                   nullptr, nullptr, 0, NC, lg);
    k_logsoftmax<<<1, 64, 0, stream>>>(lg, (float*)d_out, B, NC);
}
